// DynamicConv2D_88321707475734
// MI455X (gfx1250) — matrix operand order repaired and checked by reading, not compiled, not run on hardware
//
#include <hip/hip_runtime.h>
#include <hip/hip_bf16.h>

// ---------------------------------------------------------------------------
// DynamicConv2D with all-ones indicator => one shared conv weight/bias:
//   stage 1: tiny MLP -> h2[256], bdyn[128]
//   stage 2: wdyn[128][576] = W3 @ h2 + b3 (f16); h2 staged via TENSOR_LOAD_TO_LDS
//   stage 3: implicit-GEMM conv, v_wmma_f32_16x16x32_f16
//            - halo tile staged via GLOBAL_LOAD_ASYNC_TO_LDS_B32 (ASYNCcnt)
//            - im2col B tile repacked once in LDS, read as b128 per wave
// Shapes: B=32, C=64, H=W=96, O=128, K=3 -> GEMM M=128, K=576, N=294912
// ---------------------------------------------------------------------------

typedef _Float16 f16;
typedef __attribute__((ext_vector_type(16))) _Float16 v16h;
typedef __attribute__((ext_vector_type(8)))  _Float16 v8h;
typedef __attribute__((ext_vector_type(8)))  float    v8f;
typedef __attribute__((ext_vector_type(4)))  unsigned u32x4;
typedef __attribute__((ext_vector_type(8)))  int      i32x8;
typedef __attribute__((ext_vector_type(4)))  int      i32x4;

#define CB   32      // batch
#define CC   64      // in channels
#define CH   96
#define CW   96
#define CO   128     // out channels
#define KD   576     // C*3*3  (== 18 * 32)
#define NK   18      // K-chunks of 32
#define HW   (CH*CW)
#define HALOW 20     // padded halo width (18 used), f32
#define KDP  592     // padded im2col K stride (halves)

// ---------------- stage 1: MLP head (one block) ----------------------------
__global__ void dynconv_mlp_head(const float* __restrict__ W1, const float* __restrict__ b1,
                                 const float* __restrict__ W2, const float* __restrict__ b2,
                                 const float* __restrict__ Wb, const float* __restrict__ bb,
                                 float* __restrict__ h2_out, float* __restrict__ bd_out) {
    __shared__ float h1s[128];
    int t = threadIdx.x;
    if (t < 128) {
        float s = b1[t];
        for (int c = 0; c < CC; ++c) s += W1[t * CC + c];   // ones @ W1^T
        h1s[t] = s > 0.f ? s : 0.f;
    }
    __syncthreads();
    {
        float s = b2[t];
        for (int i = 0; i < 128; ++i) s += W2[t * 128 + i] * h1s[i];
        h2_out[t] = s > 0.f ? s : 0.f;
    }
    if (t < 128) {
        float s = bb[t];
        for (int c = 0; c < CC; ++c) s += Wb[t * CC + c];
        bd_out[t] = s;
    }
}

// ---------------- stage 2: weight generator -> f16 [O][576] ----------------
// h2 (1 KB, shared by every thread in the block) is staged into LDS through
// the Tensor Data Mover: one 1-D descriptor, completion via TENSORcnt.
__global__ void dynconv_wgen(const float* __restrict__ W3, const float* __restrict__ b3,
                             const float* __restrict__ h2, f16* __restrict__ wf) {
    __shared__ __align__(16) float h2s[256];
    int t = threadIdx.x;
#if __has_builtin(__builtin_amdgcn_tensor_load_to_lds)
    if (t < 32) {                       // wave 0 drives the DMA (EXEC ignored by TDM)
        unsigned long long ga = (unsigned long long)h2;
        u32x4 g0 = {};
        g0[0] = 1u;                                          // count=1, user descriptor
        g0[1] = (unsigned)(size_t)&h2s[0];                   // lds_addr (bits 63:32)
        g0[2] = (unsigned)ga;                                // global_addr[31:0]
        g0[3] = (unsigned)((ga >> 32) & 0x01FFFFFFu) | (2u << 30); // addr[56:32] | type=2
        i32x8 g1 = {};
        g1[0] = (2 << 16);                                   // data_size = 4 bytes
        g1[1] = (256 << 16);                                 // tensor_dim0 = 256 (bits 79:48)
        g1[2] = (1 << 16);                                   // tensor_dim1 = 1  (bits 111:80)
        g1[3] = (256 << 16);                                 // tile_dim0 = 256  (bits 127:112)
        g1[5] = 256;                                         // tensor_dim0_stride (bits 207:160)
        i32x4 z4 = {};
        i32x8 z8 = {};
        __builtin_amdgcn_tensor_load_to_lds(g0, g1, z4, z4, z8, 0);  // 6-arg form
#if __has_builtin(__builtin_amdgcn_s_wait_tensorcnt)
        __builtin_amdgcn_s_wait_tensorcnt((short)0);
#else
        asm volatile("s_wait_tensorcnt 0x0" ::: "memory");
#endif
    }
#else
    h2s[t] = h2[t];
#endif
    __syncthreads();
    int idx = blockIdx.x * 256 + t;            // 0..73727 == o*576 + c*9 + ky*3 + kx
    float s = b3[idx];
    const float* row = W3 + (long)idx * 256;
    #pragma unroll 8
    for (int i = 0; i < 256; ++i) s += row[i] * h2s[i];
    wf[idx] = (f16)s;
}

// ---- gfx1250 async global->LDS copy (VGLOBAL async encoding, ASYNCcnt) ----
__device__ __forceinline__ void async_ld_f32(const float* g, unsigned lds_off) {
    asm volatile("global_load_async_to_lds_b32 %0, %1, off"
                 :: "v"(lds_off), "v"((unsigned long long)g) : "memory");
}
__device__ __forceinline__ void wait_asynccnt0() {
    asm volatile("s_wait_asynccnt 0x0" ::: "memory");
}

// ---------------- stage 3: implicit-GEMM conv via WMMA ---------------------
// One workgroup (8 waves) per 16-pixel segment of one output row.
// Wave w computes output channels M = 16w .. 16w+15 for those 16 pixels.
__global__ void __launch_bounds__(256)
dynconv_wmma(const float* __restrict__ x, const f16* __restrict__ wf,
             const float* __restrict__ bd, float* __restrict__ out) {
    __shared__ __align__(16) float halo[CC * 3 * HALOW];   // 15360 B, f32 halo
    __shared__ __align__(32) f16   bt[16 * KDP];           // 18944 B, im2col B tile

    const int n0  = blockIdx.x * 16;
    const int b   = n0 / HW;
    const int rem = n0 - b * HW;
    const int y   = rem / CW;
    const int x0  = rem - y * CW;              // multiple of 16; tile stays in one row

    // ---- async-stage x halo: 64ch x 3rows x 18cols f32; zeros for padding
    for (int idx = threadIdx.x; idx < CC * 3 * 18; idx += 256) {
        int c   = idx / 54;
        int r   = (idx / 18) % 3;
        int col = idx % 18;
        int gy  = y + r - 1;
        int gx  = x0 + col - 1;
        float* dst = &halo[(c * 3 + r) * HALOW + col];
        if (gy >= 0 && gy < CH && gx >= 0 && gx < CW) {
            async_ld_f32(&x[((long)(b * CC + c) * CH + gy) * CW + gx],
                         (unsigned)(size_t)dst);           // LDS addr = low 32 bits
        } else {
            *dst = 0.f;                                     // OOB -> zero via ds store
        }
    }
    wait_asynccnt0();
    __syncthreads();

    // ---- repack once: im2col B tile in WMMA layout, K-contiguous per pixel
    for (int i = threadIdx.x; i < 16 * KD; i += 256) {
        int n  = i / KD;
        int k  = i - n * KD;                   // c*9 + ky*3 + kx
        int c  = k / 9;
        int r  = k - c * 9;
        int ky = r / 3;
        int kx = r - ky * 3;
        bt[n * KDP + k] = (f16)halo[(c * 3 + ky) * HALOW + n + kx];
    }
    __syncthreads();

    const int wave = threadIdx.x >> 5;
    const int lane = threadIdx.x & 31;
    const int lm   = lane & 15;
    const int grp  = lane >> 4;
    const int m0   = wave * 16;

    v8f acc = {};
    const f16* warow = wf + (m0 + lm) * KD + grp * 16;     // A: K-contiguous, k = 16*grp + s
    const f16* brow  = bt + lm * KDP + grp * 16;           // B: K-contiguous, k = 16*grp + s, 32B aligned

    #pragma unroll
    for (int t = 0; t < NK; ++t) {
        const int kk = t * 32;
        v8h a0 = *(const v8h*)(warow + kk);
        v8h a1 = *(const v8h*)(warow + kk + 8);
        if (t + 1 < NK)
            __builtin_prefetch((const void*)(warow + kk + 32), 0, 1);
        v16h a;
        #pragma unroll
        for (int i = 0; i < 8; ++i) { a[i] = a0[i]; a[i + 8] = a1[i]; }

        v16h bf = *(const v16h*)(brow + kk);               // 2x ds_load_b128

        acc = __builtin_amdgcn_wmma_f32_16x16x32_f16(
            false, a, false, bf, (short)0, acc, false, false);
    }

    // ---- epilogue: C layout -> VGPR v holds M = m0 + grp*8 + v, N = lm
    #pragma unroll
    for (int v = 0; v < 8; ++v) {
        int M = m0 + grp * 8 + v;
        out[((long)(b * CO + M) * CH + y) * CW + x0 + lm] = acc[v] + bd[M];
    }
}

// ---------------------------------------------------------------------------
extern "C" void kernel_launch(void* const* d_in, const int* in_sizes, int n_in,
                              void* d_out, int out_size, void* d_ws, size_t ws_size,
                              hipStream_t stream) {
    const float* x  = (const float*)d_in[0];
    const float* W1 = (const float*)d_in[1];
    const float* b1 = (const float*)d_in[2];
    const float* W2 = (const float*)d_in[3];
    const float* b2 = (const float*)d_in[4];
    const float* W3 = (const float*)d_in[5];
    const float* b3 = (const float*)d_in[6];
    const float* Wb = (const float*)d_in[7];
    const float* bb = (const float*)d_in[8];
    float* out = (float*)d_out;

    float* h2 = (float*)d_ws;                          // 256 f32
    float* bd = h2 + 256;                              // 128 f32
    f16*   wf = (f16*)((char*)d_ws + 2048);            // 128*576 f16 (147456 B)

    dynconv_mlp_head<<<1, 256, 0, stream>>>(W1, b1, W2, b2, Wb, bb, h2, bd);
    dynconv_wgen<<<(CO * KD) / 256, 256, 0, stream>>>(W3, b3, h2, wf);
    dynconv_wmma<<<(CB * HW) / 16, 256, 0, stream>>>(x, wf, bd, out);
}
